// SelfAttention_50414326120979
// MI455X (gfx1250) — matrix operand order repaired and checked by reading, not compiled, not run on hardware
//
#include <hip/hip_runtime.h>
#include <math.h>

// ---------------------------------------------------------------------------
// Problem constants (match reference)
// ---------------------------------------------------------------------------
#define BATCH 2
#define SEQ   2048
#define DIM   2048
#define NH    32
#define NKVH  8
#define HD    64          // head dim
#define QKDIM (NH*HD)     // 2048
#define KVDIM (NKVH*HD)   // 512

typedef __bf16 bf16;
typedef __attribute__((ext_vector_type(16))) __bf16 v16bf;
typedef __attribute__((ext_vector_type(8)))  float  v8f;

#define WMMA_BF16(a, b, c) \
  __builtin_amdgcn_wmma_f32_16x16x32_bf16(false, (a), false, (b), (short)0, (c), false, false)

// ---------------------------------------------------------------------------
// CDNA5 async global->LDS copy (ASYNCcnt) — 16 bytes per lane.
// VDST VGPR carries the wave-relative LDS byte address (= low 32 bits of the
// generic shared pointer per the LDS aperture rule addr[31:0]).
// ---------------------------------------------------------------------------
__device__ __forceinline__ void async_copy_b128(void* lds_ptr, const void* gptr) {
  unsigned lds_off = (unsigned)(size_t)lds_ptr;
  asm volatile("global_load_async_to_lds_b128 %0, %1, off"
               :: "v"(lds_off), "v"(gptr) : "memory");
}
__device__ __forceinline__ void wait_async0() {
  asm volatile("s_wait_asynccnt 0x0" ::: "memory");
}

// ---------------------------------------------------------------------------
// WMMA fragment gathers (CDNA5 16-bit layouts, wave32)
//   A 16x32: lane m = lane&15, half = lane>>4, per 32b reg v:
//            k = (v<4 ? 2v : 16+2(v-4)) + 8*half  (pairs k,k+1 packed)
//   B 32x16: lane n = lane&15, half = lane>>4, per 32b reg v:
//            k = (v<4 ? 2v : 16+2(v-4)) + 8*half  (pairs k,k+1 packed)
// Each lane gathers two separate contiguous 16B chunks -> compiler fuses into b128 loads.
// ---------------------------------------------------------------------------
__device__ __forceinline__ v16bf load_afrag(const bf16* rowptr, int k0, int half) {
  union { unsigned int u[8]; v16bf v; } r;
#pragma unroll
  for (int i = 0; i < 8; ++i) {
    int k = k0 + ((i < 4) ? 2 * i : 16 + 2 * (i - 4)) + half * 8;
    r.u[i] = *(const unsigned int*)(rowptr + k);
  }
  return r.v;
}

__device__ __forceinline__ v16bf load_bfrag(const bf16* colptr, int k0, int half) {
  union { unsigned int u[8]; v16bf v; } r;
#pragma unroll
  for (int i = 0; i < 8; ++i) {
    int k = k0 + ((i < 4) ? 2 * i : 16 + 2 * (i - 4)) + half * 8;
    r.u[i] = *(const unsigned int*)(colptr + k);
  }
  return r.v;
}

// ---------------------------------------------------------------------------
// Conversions
// ---------------------------------------------------------------------------
__global__ void cvt_f2b_kernel(const float* __restrict__ in, bf16* __restrict__ out, int n) {
  int i = blockIdx.x * blockDim.x + threadIdx.x;
  if (i < n) out[i] = (bf16)in[i];
}

// w[k][n] (KxN f32) -> wT[n][k] (NxK bf16)
__global__ void cvt_wT_kernel(const float* __restrict__ w, bf16* __restrict__ wT, int K, int N) {
  int idx = blockIdx.x * blockDim.x + threadIdx.x;
  if (idx >= K * N) return;
  int n = idx % N;
  int k = idx / N;
  wT[(size_t)n * K + k] = (bf16)w[idx];
}

// ---------------------------------------------------------------------------
// WMMA GEMM: C[MxN] = A[MxK] * B[KxN], with B given transposed (Bt[NxK]).
// 256 threads = 8 waves arranged 2(M)x4(N); each wave computes a 32x32 tile
// (2x2 WMMA register blocking -> 4 fragments feed 4 WMMAs per k-step).
// Block covers 64x128 of C.
// ---------------------------------------------------------------------------
__global__ __launch_bounds__(256)
void gemm_wmma_kernel(const bf16* __restrict__ A, const bf16* __restrict__ Bt,
                      float* __restrict__ Cf, bf16* __restrict__ Cb,
                      int M, int N, int K) {
  const int lane = threadIdx.x & 31;
  const int wv   = threadIdx.x >> 5;
  const int half = lane >> 4;
  const int lm   = lane & 15;
  const int m0 = blockIdx.y * 64  + (wv >> 2) * 32;
  const int n0 = blockIdx.x * 128 + (wv & 3) * 32;

  const bf16* arow0 = A  + (size_t)(m0 + lm) * K;
  const bf16* arow1 = A  + (size_t)(m0 + 16 + lm) * K;
  const bf16* brow0 = Bt + (size_t)(n0 + lm) * K;
  const bf16* brow1 = Bt + (size_t)(n0 + 16 + lm) * K;

  v8f acc00 = {}, acc01 = {}, acc10 = {}, acc11 = {};
  for (int k0 = 0; k0 < K; k0 += 32) {
    v16bf a0 = load_afrag(arow0, k0, half);
    v16bf a1 = load_afrag(arow1, k0, half);
    v16bf b0 = load_bfrag(brow0, k0, half);
    v16bf b1 = load_bfrag(brow1, k0, half);
    acc00 = WMMA_BF16(a0, b0, acc00);
    acc01 = WMMA_BF16(a0, b1, acc01);
    acc10 = WMMA_BF16(a1, b0, acc10);
    acc11 = WMMA_BF16(a1, b1, acc11);
  }

#pragma unroll
  for (int r = 0; r < 8; ++r) {
    int mA = m0 + r + half * 8;         // C/D layout: lanes 0-15 M=r, 16-31 M=r+8
    int mB = mA + 16;
    int nA = n0 + lm;
    int nB = nA + 16;
    if (Cf) {
      Cf[(size_t)mA * N + nA] = acc00[r];
      Cf[(size_t)mA * N + nB] = acc01[r];
      Cf[(size_t)mB * N + nA] = acc10[r];
      Cf[(size_t)mB * N + nB] = acc11[r];
    } else {
      Cb[(size_t)mA * N + nA] = (bf16)acc00[r];
      Cb[(size_t)mA * N + nB] = (bf16)acc01[r];
      Cb[(size_t)mB * N + nA] = (bf16)acc10[r];
      Cb[(size_t)mB * N + nB] = (bf16)acc11[r];
    }
  }
}

// ---------------------------------------------------------------------------
// RoPE, in place on bf16 [B*S, nheads, 64] (interleaved-pair convention)
// ---------------------------------------------------------------------------
__global__ void rope_kernel(bf16* __restrict__ x, const float* __restrict__ fc,
                            const float* __restrict__ fs, int nheads, int total) {
  int idx = blockIdx.x * blockDim.x + threadIdx.x;
  if (idx >= total) return;
  int p  = idx & 31;              // freq pair index
  int t  = idx >> 5;              // (b*S+s)*nheads + h
  int h  = t % nheads;
  int bs = t / nheads;
  int s  = bs & (SEQ - 1);
  bf16* ptr = x + ((size_t)bs * nheads + h) * HD + 2 * p;
  float c  = fc[s * 32 + p];
  float sn = fs[s * 32 + p];
  float x0 = (float)ptr[0];
  float x1 = (float)ptr[1];
  ptr[0] = (bf16)(x0 * c - x1 * sn);
  ptr[1] = (bf16)(x0 * sn + x1 * c);
}

// ---------------------------------------------------------------------------
// lambda scalar: scale = 1 - (exp(q1.k1) - exp(q2.k2) + LAMBDA_INIT)
// ---------------------------------------------------------------------------
__global__ void lambda_kernel(const float* q1, const float* k1,
                              const float* q2, const float* k2, float* out) {
  if (threadIdx.x == 0 && blockIdx.x == 0) {
    float d1 = 0.f, d2 = 0.f;
    for (int i = 0; i < HD; ++i) { d1 += q1[i] * k1[i]; d2 += q2[i] * k2[i]; }
    float lambda_init = 0.8f - 0.6f * __expf(-0.3f * 12.0f);
    float sl = __expf(d1) - __expf(d2) + lambda_init;
    *out = 1.0f - sl;
  }
}

// ---------------------------------------------------------------------------
// V transpose: vb [b][s][kvh][64] -> vt [b][kvh][64][s]
// ---------------------------------------------------------------------------
__global__ void vtrans_kernel(const bf16* __restrict__ vb, bf16* __restrict__ vt, int total) {
  int idx = blockIdx.x * blockDim.x + threadIdx.x;
  if (idx >= total) return;
  int d   = idx & 63;
  int t   = idx >> 6;
  int kvh = t & (NKVH - 1);
  int bs  = t >> 3;
  int s   = bs & (SEQ - 1);
  int b   = bs >> 11;
  vt[(((size_t)(b * NKVH + kvh) * HD + d) * SEQ) + s] = vb[idx];
}

// ---------------------------------------------------------------------------
// Flash attention (non-causal, GQA 4:1).
// Block = 128 threads = 4 waves, all on the same (b, head), covering 64
// consecutive queries (one 16-query tile per wave). K/V tiles for 32 keys are
// staged cooperatively into LDS with async global->LDS b128 copies, double
// buffered (compute tile i while tile i+1 streams in). Q.K^T and P.V run on
// v_wmma_f32_16x16x32_bf16; online softmax staged through per-wave LDS.
// ---------------------------------------------------------------------------
__global__ __launch_bounds__(128)
void attn_kernel(const bf16* __restrict__ qb, const bf16* __restrict__ kb,
                 const bf16* __restrict__ vt, bf16* __restrict__ ao,
                 const float* __restrict__ scaleptr) {
  __shared__ __attribute__((aligned(16))) bf16 Kbuf[2][32][HD];   // [key][d]
  __shared__ __attribute__((aligned(16))) bf16 Vbuf[2][HD][32];   // [d][key]
  __shared__ float ldsS[4][16 * 32];
  __shared__ __attribute__((aligned(16))) bf16 ldsP[4][16 * 32];
  __shared__ float ldsCorr[4][16];
  __shared__ float ldsRow[4][16];

  const int tid  = threadIdx.x;
  const int lane = tid & 31;
  const int wv   = tid >> 5;
  const int half = lane >> 4;
  const int lm   = lane & 15;

  const int qblk = blockIdx.x & 31;            // 64-query block within head
  const int h    = (blockIdx.x >> 5) & (NH - 1);
  const int b    = blockIdx.x >> 10;
  const int kvh  = h >> 2;                     // N_REP = 4
  const int q0   = qblk * 64 + wv * 16;

  const bf16* kbase = kb + ((size_t)b * SEQ * KVDIM + kvh * HD);
  const bf16* vbase = vt + ((size_t)(b * NKVH + kvh) * HD) * SEQ;

  // Q fragments (row = query, k = d), kept in registers for the whole pass
  const bf16* qrow = qb + ((size_t)(b * SEQ + q0 + lm) * QKDIM + h * HD);
  v16bf aq0 = load_afrag(qrow, 0, half);
  v16bf aq1 = load_afrag(qrow, 32, half);

  v8f o[4] = {};
  float runmax = -1e30f, runsum = 0.f;
  const float sscale = 0.125f;                 // 1/sqrt(64)

  // ---- cooperative async stage of one 32-key K/V tile ----
  auto stage_tile = [&](int buf, int kt) {
    // K tile: 32 rows x 64 bf16 = 256 x 16B segments; 2 per thread
#pragma unroll
    for (int s = tid; s < 256; s += 128) {
      int row = s >> 3, c = (s & 7) * 8;
      async_copy_b128(&Kbuf[buf][row][c], kbase + (size_t)(kt + row) * KVDIM + c);
    }
    // V tile: 64 d-rows x 32 bf16 = 256 x 16B segments; 2 per thread
#pragma unroll
    for (int s = tid; s < 256; s += 128) {
      int d = s >> 2, c = (s & 3) * 8;
      async_copy_b128(&Vbuf[buf][d][c], vbase + (size_t)d * SEQ + kt + c);
    }
  };

  stage_tile(0, 0);

  int buf = 0;
  for (int kt = 0; kt < SEQ; kt += 32, buf ^= 1) {
    wait_async0();          // my async copies of Kbuf/Vbuf[buf] landed
    __syncthreads();        // everyone's copies landed; prev reads of buf^1 done
    if (kt + 32 < SEQ) stage_tile(buf ^ 1, kt + 32);

    // ---- scores: two 16x16 key sub-tiles, k-dim 64 = 2 WMMA steps each ----
    v8f s0 = {}, s1 = {};
    {
      const bf16* krow0 = &Kbuf[buf][lm][0];
      const bf16* krow1 = &Kbuf[buf][16 + lm][0];
      s0 = WMMA_BF16(aq0, load_bfrag(krow0, 0, half), s0);
      s0 = WMMA_BF16(aq1, load_bfrag(krow0, 32, half), s0);
      s1 = WMMA_BF16(aq0, load_bfrag(krow1, 0, half), s1);
      s1 = WMMA_BF16(aq1, load_bfrag(krow1, 32, half), s1);
    }

#pragma unroll
    for (int r = 0; r < 8; ++r) {
      int m = r + half * 8;
      ldsS[wv][m * 32 + lm]      = s0[r] * sscale;
      ldsS[wv][m * 32 + 16 + lm] = s1[r] * sscale;
    }
    __syncthreads();

    // ---- online softmax: lane m owns query row m ----
    if (lane < 16) {
      const int m = lane;
      float row[32];
      float tmax = -1e30f;
#pragma unroll
      for (int j = 0; j < 32; ++j) { row[j] = ldsS[wv][m * 32 + j]; tmax = fmaxf(tmax, row[j]); }
      float nmax = fmaxf(runmax, tmax);
      float corr = __expf(runmax - nmax);
      float s = 0.f;
#pragma unroll
      for (int j = 0; j < 32; ++j) {
        float p = __expf(row[j] - nmax);
        s += p;
        ldsP[wv][m * 32 + j] = (bf16)p;
      }
      runsum = runsum * corr + s;
      runmax = nmax;
      ldsCorr[wv][m] = corr;
    }
    __syncthreads();

    // ---- rescale accumulators by per-row correction ----
#pragma unroll
    for (int r = 0; r < 8; ++r) {
      float c = ldsCorr[wv][r + half * 8];
#pragma unroll
      for (int cch = 0; cch < 4; ++cch) o[cch][r] *= c;
    }

    // ---- reload P as A fragment (16 queries x 32 keys) ----
    v16bf ap;
    {
      union { unsigned int u[8]; v16bf v; } t;
#pragma unroll
      for (int i = 0; i < 8; ++i) {
        int k = ((i < 4) ? 2 * i : 16 + 2 * (i - 4)) + half * 8;
        t.u[i] = *(const unsigned int*)&ldsP[wv][lm * 32 + k];
      }
      ap = t.v;
    }

    // ---- P @ V: 4 d-chunks of 16 ----
#pragma unroll
    for (int c = 0; c < 4; ++c) {
      v16bf bv = load_bfrag(&Vbuf[buf][c * 16 + lm][0], 0, half);
      o[c] = WMMA_BF16(ap, bv, o[c]);
    }
    // loop-top __syncthreads protects ldsS/ldsP/K/V buffers
  }

  // ---- epilogue: fold 1/rowsum and lambda scale, write bf16 [b][s][h][d] ----
  if (lane < 16) ldsRow[wv][lane] = (*scaleptr) / runsum;
  __syncthreads();

  bf16* orow = ao + ((size_t)(b * SEQ + q0) * QKDIM + h * HD);
#pragma unroll
  for (int r = 0; r < 8; ++r) {
    int m = r + half * 8;
    float fs = ldsRow[wv][m];
#pragma unroll
    for (int c = 0; c < 4; ++c)
      orow[(size_t)m * QKDIM + c * 16 + lm] = (bf16)(o[c][r] * fs);
  }
}

// ---------------------------------------------------------------------------
// Host side
// ---------------------------------------------------------------------------
extern "C" void kernel_launch(void* const* d_in, const int* in_sizes, int n_in,
                              void* d_out, int out_size, void* d_ws, size_t ws_size,
                              hipStream_t stream) {
  const float* x_f  = (const float*)d_in[0];
  const float* fcos = (const float*)d_in[2];
  const float* fsin = (const float*)d_in[3];
  const float* wq_f = (const float*)d_in[4];
  const float* wk_f = (const float*)d_in[5];
  const float* wv_f = (const float*)d_in[6];
  const float* wo_f = (const float*)d_in[7];
  const float* lq1  = (const float*)d_in[8];
  const float* lk1  = (const float*)d_in[9];
  const float* lq2  = (const float*)d_in[10];
  const float* lk2  = (const float*)d_in[11];
  float* out = (float*)d_out;

  const int M = BATCH * SEQ;     // 4096

  // workspace carve-up
  char* base = (char*)d_ws;
  size_t off = 0;
  auto carve = [&](size_t bytes) -> char* {
    char* p = base + off;
    off += (bytes + 255) & ~(size_t)255;
    return p;
  };
  bf16* xb   = (bf16*)carve((size_t)M * DIM * 2);
  bf16* wqT  = (bf16*)carve((size_t)QKDIM * DIM * 2);
  bf16* wkT  = (bf16*)carve((size_t)KVDIM * DIM * 2);
  bf16* wvT  = (bf16*)carve((size_t)KVDIM * DIM * 2);
  bf16* woT  = (bf16*)carve((size_t)DIM * QKDIM * 2);
  bf16* qbuf = (bf16*)carve((size_t)M * QKDIM * 2);
  bf16* kbuf = (bf16*)carve((size_t)M * KVDIM * 2);
  bf16* vbuf = (bf16*)carve((size_t)M * KVDIM * 2);
  bf16* vtb  = (bf16*)carve((size_t)M * KVDIM * 2);
  bf16* aob  = (bf16*)carve((size_t)M * QKDIM * 2);
  float* lam = (float*)carve(256);

  // 1) conversions
  {
    int n = M * DIM;
    cvt_f2b_kernel<<<(n + 255) / 256, 256, 0, stream>>>(x_f, xb, n);
  }
  cvt_wT_kernel<<<(DIM * QKDIM + 255) / 256, 256, 0, stream>>>(wq_f, wqT, DIM, QKDIM);
  cvt_wT_kernel<<<(DIM * KVDIM + 255) / 256, 256, 0, stream>>>(wk_f, wkT, DIM, KVDIM);
  cvt_wT_kernel<<<(DIM * KVDIM + 255) / 256, 256, 0, stream>>>(wv_f, wvT, DIM, KVDIM);
  cvt_wT_kernel<<<(QKDIM * DIM + 255) / 256, 256, 0, stream>>>(wo_f, woT, QKDIM, DIM);

  // 2) QKV projections (WMMA)
  gemm_wmma_kernel<<<dim3(QKDIM / 128, M / 64), 256, 0, stream>>>(
      xb, wqT, nullptr, qbuf, M, QKDIM, DIM);
  gemm_wmma_kernel<<<dim3(KVDIM / 128, M / 64), 256, 0, stream>>>(
      xb, wkT, nullptr, kbuf, M, KVDIM, DIM);
  gemm_wmma_kernel<<<dim3(KVDIM / 128, M / 64), 256, 0, stream>>>(
      xb, wvT, nullptr, vbuf, M, KVDIM, DIM);

  // 3) RoPE on Q and K (in place)
  {
    int nq = M * NH * 32;
    rope_kernel<<<(nq + 255) / 256, 256, 0, stream>>>(qbuf, fcos, fsin, NH, nq);
    int nk = M * NKVH * 32;
    rope_kernel<<<(nk + 255) / 256, 256, 0, stream>>>(kbuf, fcos, fsin, NKVH, nk);
  }

  // 4) lambda scalar + V transpose
  lambda_kernel<<<1, 64, 0, stream>>>(lq1, lk1, lq2, lk2, lam);
  {
    int nv = M * KVDIM;
    vtrans_kernel<<<(nv + 255) / 256, 256, 0, stream>>>(vbuf, vtb, nv);
  }

  // 5) flash attention (WMMA + async LDS staging)
  attn_kernel<<<BATCH * NH * (SEQ / 64), 128, 0, stream>>>(
      qbuf, kbuf, vtb, aob, lam);

  // 6) output projection (WMMA, f32 out)
  gemm_wmma_kernel<<<dim3(DIM / 128, M / 64), 256, 0, stream>>>(
      aob, woT, out, nullptr, M, DIM, QKDIM);
}
